// BioSleepXSeq_90237262889285
// MI455X (gfx1250) — compile-verified, matrix operand order checked by reading, not run on hardware
//
#include <hip/hip_runtime.h>

// ---------------------------------------------------------------------------
// BioSleepXSeq encoder block for MI455X (gfx1250, wave32, WMMA).
// GEMMs: one wave computes a 32x64 tile = 8 independent v_wmma_f32_16x16x32_f16
// accumulator chains per K-step (hides WMMA RAW hazard, 4x less A traffic).
// Scores kernel: native v_tanh_f32 when available (dominant cost: B*S*S*D tanh).
// ---------------------------------------------------------------------------

#define Bb   32
#define Ss   128
#define Dd   128
#define DFFf 512
#define DRr  32
#define Mm   64
#define BSn  (Bb * Ss)   // 4096 rows

typedef __attribute__((ext_vector_type(16))) _Float16 v16h;
typedef __attribute__((ext_vector_type(8)))  _Float16 h8v;
typedef __attribute__((ext_vector_type(8)))  float    v8f;

__device__ __forceinline__ float sigmoidf_(float x) { return 1.0f / (1.0f + expf(-x)); }

#if __has_builtin(__builtin_amdgcn_tanhf)
__device__ __forceinline__ float fast_tanh(float x) { return __builtin_amdgcn_tanhf(x); }
#elif __has_builtin(__builtin_amdgcn_tanh_f32)
__device__ __forceinline__ float fast_tanh(float x) { return __builtin_amdgcn_tanh_f32(x); }
#else
__device__ __forceinline__ float fast_tanh(float x) { return tanhf(x); }
#endif

// ---- weight conversion: f32 [In][Out] -> f16 [Out][In] (WMMA "Bt" layout) ----
__global__ __launch_bounds__(256) void k_convT(const float* __restrict__ W,
                                               _Float16* __restrict__ Wt,
                                               int In, int Out) {
    int tid = blockIdx.x * 256 + threadIdx.x;
    if (tid < In * Out) {
        int o = tid / In, i = tid - o * In;
        Wt[tid] = (_Float16)W[(size_t)i * Out + o];
    }
}
// ---- straight f32 -> f16 (memory matrix already [M][D] == [N][K]) ----
__global__ __launch_bounds__(256) void k_conv(const float* __restrict__ W,
                                              _Float16* __restrict__ Wt, int n) {
    int tid = blockIdx.x * 256 + threadIdx.x;
    if (tid < n) Wt[tid] = (_Float16)W[tid];
}

// ---------------------------------------------------------------------------
// Single-wave WMMA GEMM, 32x64 tile per wave (2 M-frags x 4 N-frags).
// A: f16 row-major [M][K] (lda), Bt: f16 [N][K] (ldb). Optional bias / residual
// / ReLU; stores to f32, f16, and/or batch-transposed f16 (for V^T).
// ---------------------------------------------------------------------------
__global__ __launch_bounds__(32) void k_gemm_wmma(
    const _Float16* __restrict__ A,  int lda, long sA,
    const _Float16* __restrict__ Bt, int ldb, long sB,
    const float* __restrict__ bias,
    const float* __restrict__ resid, int ldr, long sR,
    float*    __restrict__ outF, int ldoF, long sOF,
    _Float16* __restrict__ outH, int ldoH, long sOH,
    _Float16* __restrict__ outHT,        // [gm>>7][n][gm&127], 128x128 tiles
    int Kdim, int doRelu)
{
    const int l   = threadIdx.x;
    const int c16 = l & 15;
    const int hi  = l >> 4;
    const int tn  = blockIdx.x * 64;
    const int tm  = blockIdx.y * 32;
    const int bz  = blockIdx.z;

    const _Float16* arow0 = A  + (long)bz * sA + (long)(tm + c16) * lda;
    const _Float16* arow1 = arow0 + (long)16 * lda;
    const _Float16* brow0 = Bt + (long)bz * sB + (long)(tn + c16) * ldb;
    const _Float16* brow1 = brow0 + (long)16 * ldb;
    const _Float16* brow2 = brow0 + (long)32 * ldb;
    const _Float16* brow3 = brow0 + (long)48 * ldb;

    v8f acc[2][4];
#pragma unroll
    for (int mt = 0; mt < 2; ++mt)
#pragma unroll
        for (int j = 0; j < 4; ++j) acc[mt][j] = (v8f){};

    for (int k0 = 0; k0 < Kdim; k0 += 32) {
        const int o0 = k0 + hi * 8, o1 = k0 + 16 + hi * 8;
        h8v a00 = *(const h8v*)(arow0 + o0), a01 = *(const h8v*)(arow0 + o1);
        h8v a10 = *(const h8v*)(arow1 + o0), a11 = *(const h8v*)(arow1 + o1);
        h8v b00 = *(const h8v*)(brow0 + o0), b01 = *(const h8v*)(brow0 + o1);
        h8v b10 = *(const h8v*)(brow1 + o0), b11 = *(const h8v*)(brow1 + o1);
        h8v b20 = *(const h8v*)(brow2 + o0), b21 = *(const h8v*)(brow2 + o1);
        h8v b30 = *(const h8v*)(brow3 + o0), b31 = *(const h8v*)(brow3 + o1);
        v16h a[2], b[4];
        a[0] = __builtin_shufflevector(a00, a01, 0,1,2,3,4,5,6,7,8,9,10,11,12,13,14,15);
        a[1] = __builtin_shufflevector(a10, a11, 0,1,2,3,4,5,6,7,8,9,10,11,12,13,14,15);
        b[0] = __builtin_shufflevector(b00, b01, 0,1,2,3,4,5,6,7,8,9,10,11,12,13,14,15);
        b[1] = __builtin_shufflevector(b10, b11, 0,1,2,3,4,5,6,7,8,9,10,11,12,13,14,15);
        b[2] = __builtin_shufflevector(b20, b21, 0,1,2,3,4,5,6,7,8,9,10,11,12,13,14,15);
        b[3] = __builtin_shufflevector(b30, b31, 0,1,2,3,4,5,6,7,8,9,10,11,12,13,14,15);
#pragma unroll
        for (int mt = 0; mt < 2; ++mt)
#pragma unroll
            for (int j = 0; j < 4; ++j)
                acc[mt][j] = __builtin_amdgcn_wmma_f32_16x16x32_f16(
                    false, a[mt], false, b[j], (short)0, acc[mt][j], false, false);
    }

    float bb[4];
#pragma unroll
    for (int j = 0; j < 4; ++j) bb[j] = bias ? bias[tn + j * 16 + c16] : 0.0f;

#pragma unroll
    for (int mt = 0; mt < 2; ++mt) {
#pragma unroll
        for (int j = 0; j < 4; ++j) {
            const int gn = tn + j * 16 + c16;
#pragma unroll
            for (int r = 0; r < 8; ++r) {
                const int gm = tm + mt * 16 + r + 8 * hi;
                float v = acc[mt][j][r] + bb[j];
                if (resid)  v += resid[(long)bz * sR + (long)gm * ldr + gn];
                if (doRelu) v = fmaxf(v, 0.0f);
                if (outF)  outF[(long)bz * sOF + (long)gm * ldoF + gn] = v;
                if (outH)  outH[(long)bz * sOH + (long)gm * ldoH + gn] = (_Float16)v;
                if (outHT) outHT[(long)(gm >> 7) * (128 * 128) + (long)gn * 128 + (gm & 127)]
                               = (_Float16)v;
            }
        }
    }
}

// ---- custom LayerNorm (unbiased std, eps added to std) -> f16 ----
__global__ __launch_bounds__(128) void k_ln_custom(
    const float* __restrict__ in, const float* __restrict__ g,
    const float* __restrict__ b, _Float16* __restrict__ outH)
{
    __shared__ float red[128];
    const int row = blockIdx.x, t = threadIdx.x;
    float v = in[(long)row * 128 + t];
    red[t] = v; __syncthreads();
    for (int s = 64; s > 0; s >>= 1) { if (t < s) red[t] += red[t + s]; __syncthreads(); }
    float mean = red[0] * (1.0f / 128.0f); __syncthreads();
    float d = v - mean;
    red[t] = d * d; __syncthreads();
    for (int s = 64; s > 0; s >>= 1) { if (t < s) red[t] += red[t + s]; __syncthreads(); }
    float sd = sqrtf(red[0] * (1.0f / 127.0f));
    outH[(long)row * 128 + t] = (_Float16)(g[t] * d / (sd + 1e-6f) + b[t]);
}

// ---- additive-attention scores + softmax (the tanh hot loop) ----
__global__ __launch_bounds__(128) void k_attn_scores(
    const float* __restrict__ Q, const float* __restrict__ Km,
    const float* __restrict__ wd, const float* __restrict__ ws,
    const float* __restrict__ wt, const float* __restrict__ bd,
    const float* __restrict__ bsg, const float* __restrict__ bt,
    _Float16* __restrict__ probs)
{
    __shared__ float qrow[128], coef[128], red[128];
    const int q = blockIdx.x, b = blockIdx.y, t = threadIdx.x;
    qrow[t] = Q[((long)b * 128 + q) * 128 + t];
    coef[t] = 0.08838834764831845f /* 1/sqrt(128) */ + wd[t] + ws[t] + wt[t];
    __syncthreads();
    float acc = bd[0] + bsg[0] + bt[0];
    const float4* krow4 = (const float4*)(Km + ((long)b * 128 + t) * 128);
#pragma unroll 4
    for (int d4 = 0; d4 < 32; ++d4) {
        float4 kk = krow4[d4];
        const int d = d4 * 4;
        acc += fast_tanh(qrow[d + 0] + kk.x) * coef[d + 0];
        acc += fast_tanh(qrow[d + 1] + kk.y) * coef[d + 1];
        acc += fast_tanh(qrow[d + 2] + kk.z) * coef[d + 2];
        acc += fast_tanh(qrow[d + 3] + kk.w) * coef[d + 3];
    }
    red[t] = acc; __syncthreads();
    for (int s = 64; s > 0; s >>= 1) { if (t < s) red[t] = fmaxf(red[t], red[t + s]); __syncthreads(); }
    float mx = red[0]; __syncthreads();
    float e = expf(acc - mx);
    red[t] = e; __syncthreads();
    for (int s = 64; s > 0; s >>= 1) { if (t < s) red[t] += red[t + s]; __syncthreads(); }
    probs[((long)b * 128 + q) * 128 + t] = (_Float16)(e / red[0]);
}

// ---- channel mean over S: avg[b][d] ----
__global__ __launch_bounds__(128) void k_avg(const float* __restrict__ h2,
                                             float* __restrict__ avg) {
    const int b = blockIdx.x, d = threadIdx.x;
    float s = 0.f;
    for (int ss = 0; ss < 128; ++ss) s += h2[((long)b * 128 + ss) * 128 + d];
    avg[b * 128 + d] = s * (1.0f / 128.0f);
}

// ---- channel gate + feature gate (two tiny MLPs on avg) ----
__global__ __launch_bounds__(128) void k_gates_bd(
    const float* __restrict__ avg,
    const float* __restrict__ cgW1, const float* __restrict__ cgb1,
    const float* __restrict__ cgW2, const float* __restrict__ cgb2,
    const float* __restrict__ sqW1, const float* __restrict__ sqb1,
    const float* __restrict__ sqW2, const float* __restrict__ sqb2,
    float* __restrict__ ca, float* __restrict__ fw)
{
    __shared__ float av[128], h1[32], h2s[32];
    const int b = blockIdx.x, t = threadIdx.x;
    av[t] = avg[b * 128 + t]; __syncthreads();
    if (t < 32) {
        float a1 = cgb1[t], a2 = sqb1[t];
        for (int d = 0; d < 128; ++d) {
            a1 += av[d] * cgW1[t * 128 + d];   // cg_W1 stored [out][in]
            a2 += av[d] * sqW1[d * 32 + t];    // sq_W1 stored [in][out]
        }
        h1[t] = fmaxf(a1, 0.f); h2s[t] = fmaxf(a2, 0.f);
    }
    __syncthreads();
    float c = cgb2[t], f = sqb2[t];
    for (int r = 0; r < 32; ++r) {
        c += h1[r]  * cgW2[t * 32 + r];        // cg_W2 stored [out][in]
        f += h2s[r] * sqW2[r * 128 + t];       // sq_W2 stored [in][out]
    }
    ca[b * 128 + t] = sigmoidf_(c);
    fw[b * 128 + t] = sigmoidf_(f);
}

// ---- spatial gate: conv1x1 -> BN(eval) -> relu -> conv1x1 -> sigmoid ----
__global__ __launch_bounds__(128) void k_spatial(
    const float* __restrict__ h2,
    const float* __restrict__ sgW1, const float* __restrict__ sgb1,
    const float* __restrict__ bng, const float* __restrict__ bnb,
    const float* __restrict__ bnm, const float* __restrict__ bnv,
    const float* __restrict__ sgW2, const float* __restrict__ sgb2,
    float* __restrict__ sa)
{
    __shared__ float w1[32 * 128];
    __shared__ float sc[32], sh[32], w2[32];
    const int b = blockIdx.x, t = threadIdx.x;
    for (int i = t; i < 32 * 128; i += 128) w1[i] = sgW1[i];
    if (t < 32) {
        float inv = rsqrtf(bnv[t] + 1e-5f);
        sc[t] = inv * bng[t];
        sh[t] = bnb[t] - bnm[t] * sc[t];
        w2[t] = sgW2[t];
    }
    __syncthreads();
    float acc[32];
#pragma unroll
    for (int r = 0; r < 32; ++r) acc[r] = sgb1[r];
    const float* row = h2 + ((long)b * 128 + t) * 128;
    for (int d = 0; d < 128; ++d) {
        float x = row[d];
#pragma unroll
        for (int r = 0; r < 32; ++r) acc[r] += w1[r * 128 + d] * x;
    }
    float so = sgb2[0];
#pragma unroll
    for (int r = 0; r < 32; ++r) so += w2[r] * fmaxf(acc[r] * sc[r] + sh[r], 0.f);
    sa[b * 128 + t] = sigmoidf_(so);
}

// ---- apply gates; write g into concat[:, 0:128] as f16 ----
__global__ __launch_bounds__(128) void k_apply_gates(
    const float* __restrict__ h2, const float* __restrict__ ca,
    const float* __restrict__ sa, const float* __restrict__ fw,
    _Float16* __restrict__ concatA)
{
    const int row = blockIdx.x, t = threadIdx.x;
    const int b = row >> 7;
    float v = h2[(long)row * 128 + t] * ca[b * 128 + t] * sa[row] * fw[b * 128 + t];
    concatA[(long)row * 256 + t] = (_Float16)v;
}

// ---- top-3 over M=64 sims + softmax-weighted memory mix -> concat[:,128:256] ----
__global__ __launch_bounds__(128) void k_topk_mem(
    const float* __restrict__ sim, const float* __restrict__ mem,
    _Float16* __restrict__ concatA)
{
    __shared__ float sv[64];
    __shared__ float wv[3];
    __shared__ int   ti[3];
    const int row = blockIdx.x, t = threadIdx.x;
    if (t < 64) sv[t] = sim[(long)row * 64 + t];
    __syncthreads();
    if (t == 0) {
        float v[3];
        for (int j = 0; j < 3; ++j) {
            int bi = 0; float bvv = -3.0e38f;
            for (int i = 0; i < 64; ++i) if (sv[i] > bvv) { bvv = sv[i]; bi = i; }
            v[j] = bvv; ti[j] = bi; sv[bi] = -3.0e38f;
        }
        float e0 = 1.0f, e1 = expf(v[1] - v[0]), e2 = expf(v[2] - v[0]);
        float inv = 1.0f / (e0 + e1 + e2);
        wv[0] = e0 * inv; wv[1] = e1 * inv; wv[2] = e2 * inv;
    }
    __syncthreads();
    float mo = wv[0] * mem[ti[0] * 128 + t] + wv[1] * mem[ti[1] * 128 + t] +
               wv[2] * mem[ti[2] * 128 + t];
    concatA[(long)row * 256 + 128 + t] = (_Float16)mo;
}

// ---- final nn.LayerNorm (biased var, eps inside sqrt) + ReLU ----
__global__ __launch_bounds__(128) void k_lnf_relu(
    const float* __restrict__ y, const float* __restrict__ g,
    const float* __restrict__ b, float* __restrict__ out)
{
    __shared__ float red[128];
    const int row = blockIdx.x, t = threadIdx.x;
    float v = y[(long)row * 128 + t];
    red[t] = v; __syncthreads();
    for (int s = 64; s > 0; s >>= 1) { if (t < s) red[t] += red[t + s]; __syncthreads(); }
    float mean = red[0] * (1.0f / 128.0f); __syncthreads();
    float d = v - mean;
    red[t] = d * d; __syncthreads();
    for (int s = 64; s > 0; s >>= 1) { if (t < s) red[t] += red[t + s]; __syncthreads(); }
    float inv = rsqrtf(red[0] * (1.0f / 128.0f) + 1e-5f);
    out[(long)row * 128 + t] = fmaxf(d * inv * g[t] + b[t], 0.0f);
}

// ---------------------------------------------------------------------------
extern "C" void kernel_launch(void* const* d_in, const int* in_sizes, int n_in,
                              void* d_out, int out_size, void* d_ws, size_t ws_size,
                              hipStream_t stream) {
    (void)in_sizes; (void)n_in; (void)out_size; (void)ws_size;
    // input order (setup_inputs dict order)
    const float* x     = (const float*)d_in[0];
    const float* Wq    = (const float*)d_in[1];  const float* bq  = (const float*)d_in[2];
    const float* Wk    = (const float*)d_in[3];  const float* bk  = (const float*)d_in[4];
    const float* Wv    = (const float*)d_in[5];  const float* bv  = (const float*)d_in[6];
    const float* wdel  = (const float*)d_in[7];  const float* bdel = (const float*)d_in[8];
    const float* wsig  = (const float*)d_in[9];  const float* bsig = (const float*)d_in[10];
    const float* wthe  = (const float*)d_in[11]; const float* bthe = (const float*)d_in[12];
    const float* ln1g  = (const float*)d_in[13]; const float* ln1b = (const float*)d_in[14];
    const float* ln2g  = (const float*)d_in[15]; const float* ln2b = (const float*)d_in[16];
    const float* W1    = (const float*)d_in[17]; const float* b1  = (const float*)d_in[18];
    const float* W2    = (const float*)d_in[19]; const float* b2  = (const float*)d_in[20];
    const float* sqW1  = (const float*)d_in[21]; const float* sqb1 = (const float*)d_in[22];
    const float* sqW2  = (const float*)d_in[23]; const float* sqb2 = (const float*)d_in[24];
    const float* sgW1  = (const float*)d_in[25]; const float* sgb1 = (const float*)d_in[26];
    const float* bng   = (const float*)d_in[27]; const float* bnb = (const float*)d_in[28];
    const float* bnm   = (const float*)d_in[29]; const float* bnv = (const float*)d_in[30];
    const float* sgW2  = (const float*)d_in[31]; const float* sgb2 = (const float*)d_in[32];
    const float* cgW1  = (const float*)d_in[33]; const float* cgb1 = (const float*)d_in[34];
    const float* cgW2  = (const float*)d_in[35]; const float* cgb2 = (const float*)d_in[36];
    const float* mem   = (const float*)d_in[37];
    const float* Wqm   = (const float*)d_in[38]; const float* bqm = (const float*)d_in[39];
    const float* Wf    = (const float*)d_in[40]; const float* bf  = (const float*)d_in[41];
    const float* lnfg  = (const float*)d_in[42]; const float* lnfb = (const float*)d_in[43];
    float* out = (float*)d_out;

    // ---- workspace carve-out ----
    char* p = (char*)d_ws;
    auto alloc = [&](size_t bytes) { char* r = p; p += (bytes + 255) & ~(size_t)255; return r; };
    _Float16* WqT   = (_Float16*)alloc(128 * 128 * 2);
    _Float16* WkT   = (_Float16*)alloc(128 * 128 * 2);
    _Float16* WvT   = (_Float16*)alloc(128 * 128 * 2);
    _Float16* W1T   = (_Float16*)alloc(512 * 128 * 2);
    _Float16* W2T   = (_Float16*)alloc(128 * 512 * 2);
    _Float16* WqmT  = (_Float16*)alloc(128 * 128 * 2);
    _Float16* WfT   = (_Float16*)alloc(128 * 256 * 2);
    _Float16* memH  = (_Float16*)alloc(64 * 128 * 2);
    _Float16* lnH   = (_Float16*)alloc((size_t)BSn * 128 * 2);
    float*    Qf    = (float*)   alloc((size_t)BSn * 128 * 4);
    float*    Kf    = (float*)   alloc((size_t)BSn * 128 * 4);
    _Float16* VtH   = (_Float16*)alloc((size_t)BSn * 128 * 2);   // per-batch V^T
    _Float16* probsH= (_Float16*)alloc((size_t)BSn * 128 * 2);
    float*    hf    = (float*)   alloc((size_t)BSn * 128 * 4);
    _Float16* ffnH  = (_Float16*)alloc((size_t)BSn * 512 * 2);
    float*    h2f   = (float*)   alloc((size_t)BSn * 128 * 4);
    float*    avgB  = (float*)   alloc((size_t)Bb * 128 * 4);
    float*    caB   = (float*)   alloc((size_t)Bb * 128 * 4);
    float*    fwB   = (float*)   alloc((size_t)Bb * 128 * 4);
    float*    saB   = (float*)   alloc((size_t)Bb * 128 * 4);
    _Float16* concatH = (_Float16*)alloc((size_t)BSn * 256 * 2);
    _Float16* qmH   = (_Float16*)alloc((size_t)BSn * 128 * 2);
    float*    simF  = (float*)   alloc((size_t)BSn * 64 * 4);
    float*    yF    = (float*)   alloc((size_t)BSn * 128 * 4);

    // ---- weight conversion / transpose to f16 [N][K] ----
    k_convT<<<(128 * 128 + 255) / 256, 256, 0, stream>>>(Wq,  WqT,  128, 128);
    k_convT<<<(128 * 128 + 255) / 256, 256, 0, stream>>>(Wk,  WkT,  128, 128);
    k_convT<<<(128 * 128 + 255) / 256, 256, 0, stream>>>(Wv,  WvT,  128, 128);
    k_convT<<<(128 * 512 + 255) / 256, 256, 0, stream>>>(W1,  W1T,  128, 512);
    k_convT<<<(512 * 128 + 255) / 256, 256, 0, stream>>>(W2,  W2T,  512, 128);
    k_convT<<<(128 * 128 + 255) / 256, 256, 0, stream>>>(Wqm, WqmT, 128, 128);
    k_convT<<<(256 * 128 + 255) / 256, 256, 0, stream>>>(Wf,  WfT,  256, 128);
    k_conv <<<(64 * 128 + 255) / 256, 256, 0, stream>>>(mem, memH, 64 * 128);

    // ---- sublayer 1: LN1 + QKV projections (M=4096 -> grid.y=128; N=128 -> grid.x=2) ----
    k_ln_custom<<<BSn, 128, 0, stream>>>(x, ln1g, ln1b, lnH);
    k_gemm_wmma<<<dim3(2, 128, 1), 32, 0, stream>>>(lnH, 128, 0L, WqT, 128, 0L, bq,
        nullptr, 0, 0L, Qf, 128, 0L, nullptr, 0, 0L, nullptr, 128, 0);
    k_gemm_wmma<<<dim3(2, 128, 1), 32, 0, stream>>>(lnH, 128, 0L, WkT, 128, 0L, bk,
        nullptr, 0, 0L, Kf, 128, 0L, nullptr, 0, 0L, nullptr, 128, 0);
    k_gemm_wmma<<<dim3(2, 128, 1), 32, 0, stream>>>(lnH, 128, 0L, WvT, 128, 0L, bv,
        nullptr, 0, 0L, nullptr, 0, 0L, nullptr, 0, 0L, VtH, 128, 0);

    // ---- scores + softmax (tanh hot loop), then A@V via WMMA (batched, +x residual) ----
    k_attn_scores<<<dim3(Ss, Bb, 1), 128, 0, stream>>>(Qf, Kf, wdel, wsig, wthe,
                                                       bdel, bsig, bthe, probsH);
    k_gemm_wmma<<<dim3(2, 4, Bb), 32, 0, stream>>>(probsH, 128, (long)(128 * 128),
        VtH, 128, (long)(128 * 128), nullptr, x, 128, (long)(128 * 128),
        hf, 128, (long)(128 * 128), nullptr, 0, 0L, nullptr, 128, 0);

    // ---- sublayer 2: LN2 + FFN (relu GEMM, GEMM + residual) ----
    k_ln_custom<<<BSn, 128, 0, stream>>>(hf, ln2g, ln2b, lnH);
    k_gemm_wmma<<<dim3(8, 128, 1), 32, 0, stream>>>(lnH, 128, 0L, W1T, 128, 0L, b1,
        nullptr, 0, 0L, nullptr, 0, 0L, ffnH, 512, 0L, nullptr, 128, 1);
    k_gemm_wmma<<<dim3(2, 128, 1), 32, 0, stream>>>(ffnH, 512, 0L, W2T, 512, 0L, b2,
        hf, 128, 0L, h2f, 128, 0L, nullptr, 0, 0L, nullptr, 512, 0);

    // ---- gating module ----
    k_avg<<<Bb, 128, 0, stream>>>(h2f, avgB);
    k_gates_bd<<<Bb, 128, 0, stream>>>(avgB, cgW1, cgb1, cgW2, cgb2,
                                       sqW1, sqb1, sqW2, sqb2, caB, fwB);
    k_spatial<<<Bb, 128, 0, stream>>>(h2f, sgW1, sgb1, bng, bnb, bnm, bnv,
                                      sgW2, sgb2, saB);
    k_apply_gates<<<BSn, 128, 0, stream>>>(h2f, caB, saB, fwB, concatH);

    // ---- memory-enhanced module ----
    k_gemm_wmma<<<dim3(2, 128, 1), 32, 0, stream>>>(concatH, 256, 0L, WqmT, 128, 0L, bqm,
        nullptr, 0, 0L, nullptr, 0, 0L, qmH, 128, 0L, nullptr, 128, 0);
    k_gemm_wmma<<<dim3(1, 128, 1), 32, 0, stream>>>(qmH, 128, 0L, memH, 128, 0L, nullptr,
        nullptr, 0, 0L, simF, 64, 0L, nullptr, 0, 0L, nullptr, 128, 0);
    k_topk_mem<<<BSn, 128, 0, stream>>>(simF, mem, concatH);
    k_gemm_wmma<<<dim3(2, 128, 1), 32, 0, stream>>>(concatH, 256, 0L, WfT, 256, 0L, bf,
        nullptr, 0, 0L, yF, 128, 0L, nullptr, 0, 0L, nullptr, 256, 0);

    // ---- final LayerNorm + ReLU ----
    k_lnf_relu<<<BSn, 128, 0, stream>>>(yF, lnfg, lnfb, out);
}
